// EXPHORMER_11493332484365
// MI455X (gfx1250) — matrix operand order repaired and checked by reading, not compiled, not run on hardware
//
#include <hip/hip_runtime.h>
#include <hip/hip_bf16.h>

// ---------------- problem constants (match reference) ----------------
#define NN    20000          // N_NODES
#define NVIRT 1
#define NVTOT 20001          // NN + NVIRT
#define NVP   20016          // padded to multiple of 16 for WMMA tiles
#define D     128            // HIDDEN
#define D4    512            // 4*HIDDEN
#define NH    8              // HEADS
#define DH    16             // head dim
#define NE    320000         // N_EDGES
#define EL    (NE + 2*NN)    // local edges after VirtualNode transform = 360000
#define EX    (NVTOT*4)      // expander edges per layer = 80004
#define NLAYERS 3

// ---------------- WMMA vector types ----------------
typedef __attribute__((ext_vector_type(16))) _Float16 v16h;
typedef __attribute__((ext_vector_type(8)))  _Float16 v8h;
typedef __attribute__((ext_vector_type(8)))  float    v8f;

enum { GF_GELU = 1, GF_ACC = 2 };

// =====================================================================
// GEMM: out[M,N] = A[M,K](f16) * Bt[N,K]^T(f16) + bias[N]  (+ optional
// accumulate of prior outF32, optional exact GELU, optional f16 copy)
// One wave32 per 16x16 tile, v_wmma_f32_16x16x32_f16 over K.
// =====================================================================
__global__ __launch_bounds__(256)
void exph_gemm_wmma(const _Float16* __restrict__ A,
                    const _Float16* __restrict__ Bt,
                    const float*    __restrict__ bias,
                    float*          __restrict__ outF32,
                    _Float16*       __restrict__ outF16,
                    int M, int N, int K, int flags)
{
    const int tid   = threadIdx.x;
    const int wave  = tid >> 5;
    const int lane  = tid & 31;
    const int row0  = (blockIdx.y * 8 + wave) << 4;
    if (row0 >= M) return;                       // wave-uniform: EXEC stays full
    const int lan15 = lane & 15;
    const int hi    = lane >> 4;                 // half-wave select
    const int col   = (blockIdx.x << 4) + lan15; // C/D frag: all 8 elems share N=col

    float bv = bias ? bias[col] : 0.0f;
    v8f acc;
    if (flags & GF_ACC) {
#pragma unroll
        for (int f = 0; f < 8; ++f)
            acc[f] = bv + outF32[(size_t)(row0 + f + hi * 8) * N + col];
    } else {
#pragma unroll
        for (int f = 0; f < 8; ++f) acc[f] = bv;
    }

    const _Float16* arow = A  + (size_t)(row0 + lan15) * K;
    const _Float16* brow = Bt + (size_t)col * K;

    for (int kb = 0; kb < K; kb += 32) {
        // A 16x32 f16 layout: lanes0-15 M=lane K= kb+[0..7] / kb+16+[0..7];
        //                     lanes16-31 same M, K shifted by +8 in each chunk.
        v8h a_lo = *(const v8h*)(arow + kb + hi * 8);
        v8h a_hi = *(const v8h*)(arow + kb + 16 + hi * 8);
        v16h a;
#pragma unroll
        for (int j = 0; j < 8; ++j) { a[j] = a_lo[j]; a[j + 8] = a_hi[j]; }
        // B 32x16 f16 layout: lane gives N, same K order as A
        // (K= kb+8*hi+[0..7] / kb+16+8*hi+[0..7]).
        v8h b_lo = *(const v8h*)(brow + kb + hi * 8);
        v8h b_hi = *(const v8h*)(brow + kb + 16 + hi * 8);
        v16h b;
#pragma unroll
        for (int j = 0; j < 8; ++j) { b[j] = b_lo[j]; b[j + 8] = b_hi[j]; }
        acc = __builtin_amdgcn_wmma_f32_16x16x32_f16(false, a, false, b,
                                                     (short)0, acc, false, false);
    }

#pragma unroll
    for (int f = 0; f < 8; ++f) {
        float v = acc[f];
        if (flags & GF_GELU)
            v = 0.5f * v * (1.0f + erff(v * 0.70710678118654752f));
        size_t o = (size_t)(row0 + f + hi * 8) * N + col;
        if (outF32) outF32[o] = v;
        if (outF16) outF16[o] = (_Float16)v;
    }
}

// ---------------- weight transpose + f32->f16 convert: WT[N,K] = W[K,N] ----
__global__ void exph_xpose16(const float* __restrict__ W, _Float16* __restrict__ WT,
                             int K, int N)
{
    int i = blockIdx.x * blockDim.x + threadIdx.x;
    if (i >= K * N) return;
    int n = i / K, k = i - n * K;
    WT[i] = (_Float16)W[(size_t)k * N + n];
}

// ---------------- build padded node features (virtual node + pad = 0) ------
__global__ void exph_build_xv(const float* __restrict__ x,
                              float* __restrict__ xv, _Float16* __restrict__ xv16)
{
    int i = blockIdx.x * blockDim.x + threadIdx.x;
    if (i >= NVP * D) return;
    float v = (i < NN * D) ? x[i] : 0.0f;
    xv[i] = v; xv16[i] = (_Float16)v;
}

// ---------------- local edge list with virtual node -------------------------
__global__ void exph_build_edges(const int* __restrict__ ei,
                                 int* __restrict__ vsrc, int* __restrict__ vdst)
{
    int i = blockIdx.x * blockDim.x + threadIdx.x;
    if (i >= EL) return;
    int s, d;
    if (i < NE)            { s = ei[i];      d = ei[NE + i]; }
    else if (i < NE + NN)  { s = i - NE;     d = NN;         }
    else                   { s = NN;         d = i - NE - NN;}
    vsrc[i] = s; vdst[i] = d;
}

// ---------------- attention pass 1: e = exp(q[dst].k[src]/4), sum per (dst,h)
__global__ void exph_scores(const float* __restrict__ q, const float* __restrict__ k,
                            const int* __restrict__ src, const int* __restrict__ dst,
                            float* __restrict__ ebuf, float* __restrict__ ssum, int E)
{
    int i = blockIdx.x * blockDim.x + threadIdx.x;
    if (i >= E * NH) return;
    int e = i >> 3, h = i & 7;
    const float* qp = q + (size_t)dst[e] * D + h * DH;
    const float* kp = k + (size_t)src[e] * D + h * DH;
    float acc = 0.0f;
#pragma unroll
    for (int j = 0; j < DH; ++j) acc += qp[j] * kp[j];
    float ev = expf(acc * 0.25f);        // max-shift cancels; scores are O(1)
    ebuf[i] = ev;
    atomicAdd(&ssum[(size_t)dst[e] * NH + h], ev);
}

// ---------------- attention pass 2: out[dst] += (e/s) * v[src] --------------
__global__ void exph_aggregate(const float* __restrict__ v,
                               const int* __restrict__ src, const int* __restrict__ dst,
                               const float* __restrict__ ebuf, const float* __restrict__ ssum,
                               float* __restrict__ out, int E)
{
    int i = blockIdx.x * blockDim.x + threadIdx.x;
    if (i >= E * D) return;
    int e = i >> 7, c = i & 127, h = c >> 4;
    int d = dst[e];
    float w = ebuf[e * NH + h] / (ssum[(size_t)d * NH + h] + 1e-16f);
    atomicAdd(&out[(size_t)d * D + c], w * v[(size_t)src[e] * D + c]);
}

// ---------------- f32 -> f16 copy -------------------------------------------
__global__ void exph_cvt16(const float* __restrict__ a, _Float16* __restrict__ b, int n)
{
    int i = blockIdx.x * blockDim.x + threadIdx.x;
    if (i < n) b[i] = (_Float16)a[i];
}

// ---------------- residual + LN, one wave32 per 128-wide row ----------------
__global__ __launch_bounds__(256)
void exph_layernorm(const float* __restrict__ resid, const float* __restrict__ add,
                    const float* __restrict__ g, const float* __restrict__ b,
                    float* __restrict__ xv, _Float16* __restrict__ xv16, int M)
{
    int wave = threadIdx.x >> 5, lane = threadIdx.x & 31;
    int row = blockIdx.x * 8 + wave;
    if (row >= M) return;
    const float* rp = resid + (size_t)row * D;
    const float* ap = add   + (size_t)row * D;
    float vals[4], s = 0.0f, s2 = 0.0f;
#pragma unroll
    for (int j = 0; j < 4; ++j) {
        int c = lane * 4 + j;
        float v = rp[c] + ap[c];
        vals[j] = v; s += v; s2 += v * v;
    }
#pragma unroll
    for (int m = 16; m >= 1; m >>= 1) {
        s  += __shfl_xor(s,  m, 32);
        s2 += __shfl_xor(s2, m, 32);
    }
    float mean = s * (1.0f / D);
    float var  = s2 * (1.0f / D) - mean * mean;
    float inv  = rsqrtf(var + 1e-5f);
#pragma unroll
    for (int j = 0; j < 4; ++j) {
        int c = lane * 4 + j;
        float o = (vals[j] - mean) * inv * g[c] + b[c];
        xv[(size_t)row * D + c]   = o;
        xv16[(size_t)row * D + c] = (_Float16)o;
    }
}

// =====================================================================
// host driver
// =====================================================================
static inline int cdiv(int a, int b) { return (a + b - 1) / b; }

extern "C" void kernel_launch(void* const* d_in, const int* in_sizes, int n_in,
                              void* d_out, int out_size, void* d_ws, size_t ws_size,
                              hipStream_t stream)
{
    (void)in_sizes; (void)n_in; (void)out_size; (void)ws_size;

    const float* x    = (const float*)d_in[0];
    const int*   ei   = (const int*)d_in[1];
    const int*   expe = (const int*)d_in[2];
    const float* Wf[8], *Bf[8];   // lq lk lv lo eq ek ev eo
    for (int m = 0; m < 8; ++m) { Wf[m] = (const float*)d_in[3 + 2 * m];
                                  Bf[m] = (const float*)d_in[4 + 2 * m]; }
    const float* ln_g = (const float*)d_in[19];
    const float* ln_b = (const float*)d_in[20];
    const float* w1   = (const float*)d_in[21];
    const float* b1   = (const float*)d_in[22];
    const float* w2   = (const float*)d_in[23];
    const float* b2   = (const float*)d_in[24];

    // ---------------- workspace layout (256B aligned) ----------------
    char* wsb = (char*)d_ws;
    size_t off = 0;
    auto alloc = [&](size_t bytes) -> char* {
        char* p = wsb + off;
        off = (off + bytes + 255) & ~(size_t)255;
        return p;
    };
    float*    xv32  = (float*)   alloc((size_t)NVP * D * 4);
    _Float16* xv16  = (_Float16*)alloc((size_t)NVP * D * 2);
    float*    q32   = (float*)   alloc((size_t)NVP * D * 4);
    float*    k32   = (float*)   alloc((size_t)NVP * D * 4);
    float*    v32   = (float*)   alloc((size_t)NVP * D * 4);
    float*    agg32 = (float*)   alloc((size_t)NVP * D * 4);
    _Float16* agg16 = (_Float16*)alloc((size_t)NVP * D * 2);
    float*    sum32 = (float*)   alloc((size_t)NVP * D * 4);   // local+expander attn out
    _Float16* h116  = (_Float16*)alloc((size_t)NVP * D4 * 2);  // gelu(FFN hidden), f16
    float*    ebuf  = (float*)   alloc((size_t)EL * NH * 4);   // shared local/expander
    float*    ssum  = (float*)   alloc((size_t)NVP * NH * 4);
    int*      vsrc  = (int*)     alloc((size_t)EL * 4);
    int*      vdst  = (int*)     alloc((size_t)EL * 4);
    const int LSH   = 8 * D * D + D * D4 + D4 * D;             // halves per layer
    _Float16* wT    = (_Float16*)alloc((size_t)NLAYERS * LSH * 2);

    auto wTattn = [&](int l, int m) { return wT + (size_t)l * LSH + (size_t)m * D * D; };
    auto wTf1   = [&](int l) { return wT + (size_t)l * LSH + 8 * D * D; };
    auto wTf2   = [&](int l) { return wT + (size_t)l * LSH + 8 * D * D + D * D4; };

    // ---------------- one-time prep (deterministic, re-run every call) ------
    for (int l = 0; l < NLAYERS; ++l) {
        for (int m = 0; m < 8; ++m)
            exph_xpose16<<<cdiv(D * D, 256), 256, 0, stream>>>(
                Wf[m] + (size_t)l * D * D, wTattn(l, m), D, D);
        exph_xpose16<<<cdiv(D * D4, 256), 256, 0, stream>>>(
            w1 + (size_t)l * D * D4, wTf1(l), D, D4);
        exph_xpose16<<<cdiv(D4 * D, 256), 256, 0, stream>>>(
            w2 + (size_t)l * D4 * D, wTf2(l), D4, D);
    }
    exph_build_xv<<<cdiv(NVP * D, 256), 256, 0, stream>>>(x, xv32, xv16);
    exph_build_edges<<<cdiv(EL, 256), 256, 0, stream>>>(ei, vsrc, vdst);

    auto gemm = [&](const _Float16* A, const _Float16* Bt, const float* bias,
                    float* o32, _Float16* o16, int N, int K, int flags) {
        dim3 grid(N / 16, cdiv(NVP / 16, 8));
        exph_gemm_wmma<<<grid, 256, 0, stream>>>(A, Bt, bias, o32, o16, NVP, N, K, flags);
    };

    auto attention = [&](int l, int wbase, const int* sp, const int* dp, int E, int accf) {
        gemm(xv16, wTattn(l, wbase + 0), Bf[wbase + 0] + (size_t)l * D, q32, nullptr, D, D, 0);
        gemm(xv16, wTattn(l, wbase + 1), Bf[wbase + 1] + (size_t)l * D, k32, nullptr, D, D, 0);
        gemm(xv16, wTattn(l, wbase + 2), Bf[wbase + 2] + (size_t)l * D, v32, nullptr, D, D, 0);
        hipMemsetAsync(ssum, 0, (size_t)NVP * NH * 4, stream);
        hipMemsetAsync(agg32, 0, (size_t)NVP * D * 4, stream);
        exph_scores<<<cdiv(E * NH, 256), 256, 0, stream>>>(q32, k32, sp, dp, ebuf, ssum, E);
        exph_aggregate<<<cdiv(E * D, 256), 256, 0, stream>>>(v32, sp, dp, ebuf, ssum, agg32, E);
        exph_cvt16<<<cdiv(NVP * D, 256), 256, 0, stream>>>(agg32, agg16, NVP * D);
        gemm(agg16, wTattn(l, wbase + 3), Bf[wbase + 3] + (size_t)l * D, sum32, nullptr, D, D, accf);
    };

    // ---------------- layers ----------------
    for (int l = 0; l < NLAYERS; ++l) {
        // local attention -> sum32 ; expander attention accumulates into sum32
        attention(l, 0, vsrc, vdst, EL, 0);
        const int* esp = expe + (size_t)l * 2 * EX;
        const int* edp = esp + EX;
        attention(l, 4, esp, edp, EX, GF_ACC);

        // xv = LN(residual + local + expander)
        exph_layernorm<<<cdiv(NVP, 8), 256, 0, stream>>>(
            xv32, sum32, ln_g + (size_t)l * D, ln_b + (size_t)l * D, xv32, xv16, NVP);

        // FFN: h1 = gelu(xv @ W1 + b1) (f16 only); xv += h1 @ W2 + b2
        gemm(xv16, wTf1(l), b1 + (size_t)l * D4, nullptr, h116, D4, D, GF_GELU);
        gemm(h116, wTf2(l), b2 + (size_t)l * D,  xv32,    xv16, D,  D4, GF_ACC);
    }

    hipMemcpyAsync(d_out, xv32, (size_t)NN * D * 4, hipMemcpyDeviceToDevice, stream);
}
